// FullLayerChain_62302795596568
// MI455X (gfx1250) — matrix operand order repaired and checked by reading, not compiled, not run on hardware
//
#include <hip/hip_runtime.h>
#include <hip/hip_bf16.h>
#include <math.h>

// MI455X / gfx1250: wave32, WMMA bf16 16x16x32 with fp32 accumulate.
typedef __bf16 bf16_t;
typedef __attribute__((ext_vector_type(16))) __bf16 v16bf;
typedef __attribute__((ext_vector_type(8)))  __bf16 v8bf;
typedef __attribute__((ext_vector_type(8)))  float  v8f;

union ABfrag { v16bf v; v8bf h[2]; };

#define DMODEL 1024
#define RTOTAL 16384  // B*S = 4*4096

// ---------------------------------------------------------------------------
// Kernel 1: fp32 -> bf16 convert (grid-stride, bandwidth trivial)
// ---------------------------------------------------------------------------
__global__ void cvt_f32_bf16(const float* __restrict__ src,
                             bf16_t* __restrict__ dst, int n) {
    int i = blockIdx.x * blockDim.x + threadIdx.x;
    int stride = gridDim.x * blockDim.x;
    for (; i < n; i += stride) dst[i] = (bf16_t)src[i];
}

// ---------------------------------------------------------------------------
// Kernel 2: V = X*Wv^T, G = X*Wg^T, gated = V * silu(G), stored bf16.
// Block = 256 threads = 8 waves, tile 128(M) x 64(N); waves arranged 4(M)x2(N).
// Each wave: 32x32 for BOTH V and G -> 8 acc tiles, 8 WMMAs per 12 b128 loads.
// ---------------------------------------------------------------------------
__global__ __launch_bounds__(256) void proj_gate_kernel(
    const bf16_t* __restrict__ Xb,   // [RTOTAL, 1024]
    const bf16_t* __restrict__ Wv,   // [1024, 1024] row-major [out,in]
    const bf16_t* __restrict__ Wg,   // [1024, 1024]
    bf16_t* __restrict__ Gout)       // [RTOTAL, 1024] gated, bf16
{
    const int lane = threadIdx.x & 31;
    const int wave = threadIdx.x >> 5;
    const int h    = lane >> 4;   // half-wave select
    const int l15  = lane & 15;

    const int mW = blockIdx.y * 128 + (wave & 3) * 32;   // 32 rows per wave
    const int nW = blockIdx.x * 64  + (wave >> 2) * 32;  // 32 cols per wave

    const bf16_t* xrow0 = Xb + (size_t)(mW + l15) * DMODEL;       // A rows 0-15
    const bf16_t* xrow1 = Xb + (size_t)(mW + 16 + l15) * DMODEL;  // A rows 16-31
    const bf16_t* vrow0 = Wv + (size_t)(nW + l15) * DMODEL;       // B cols 0-15
    const bf16_t* vrow1 = Wv + (size_t)(nW + 16 + l15) * DMODEL;  // B cols 16-31
    const bf16_t* grow0 = Wg + (size_t)(nW + l15) * DMODEL;
    const bf16_t* grow1 = Wg + (size_t)(nW + 16 + l15) * DMODEL;

    v8f zero = {0.f,0.f,0.f,0.f,0.f,0.f,0.f,0.f};
    v8f cv[2][2], cg[2][2];
    #pragma unroll
    for (int m = 0; m < 2; ++m)
        #pragma unroll
        for (int n = 0; n < 2; ++n) { cv[m][n] = zero; cg[m][n] = zero; }

    for (int k0 = 0; k0 < DMODEL; k0 += 32) {
        ABfrag a0, a1, bv0, bv1, bg0, bg1;
        // A 16x32 bf16 layout: 16 contiguous K at K = 16h
        a0.h[0] = *(const v8bf*)(xrow0 + k0 + 16 * h);
        a0.h[1] = *(const v8bf*)(xrow0 + k0 + 16 * h + 8);
        a1.h[0] = *(const v8bf*)(xrow1 + k0 + 16 * h);
        a1.h[1] = *(const v8bf*)(xrow1 + k0 + 16 * h + 8);
        // B 32x16 bf16 layout: 16 contiguous K at K = 16h
        bv0.h[0] = *(const v8bf*)(vrow0 + k0 + 16 * h);
        bv0.h[1] = *(const v8bf*)(vrow0 + k0 + 16 * h + 8);
        bv1.h[0] = *(const v8bf*)(vrow1 + k0 + 16 * h);
        bv1.h[1] = *(const v8bf*)(vrow1 + k0 + 16 * h + 8);
        bg0.h[0] = *(const v8bf*)(grow0 + k0 + 16 * h);
        bg0.h[1] = *(const v8bf*)(grow0 + k0 + 16 * h + 8);
        bg1.h[0] = *(const v8bf*)(grow1 + k0 + 16 * h);
        bg1.h[1] = *(const v8bf*)(grow1 + k0 + 16 * h + 8);

        cv[0][0] = __builtin_amdgcn_wmma_f32_16x16x32_bf16(false, a0.v, false, bv0.v, (short)0, cv[0][0], false, false);
        cv[0][1] = __builtin_amdgcn_wmma_f32_16x16x32_bf16(false, a0.v, false, bv1.v, (short)0, cv[0][1], false, false);
        cv[1][0] = __builtin_amdgcn_wmma_f32_16x16x32_bf16(false, a1.v, false, bv0.v, (short)0, cv[1][0], false, false);
        cv[1][1] = __builtin_amdgcn_wmma_f32_16x16x32_bf16(false, a1.v, false, bv1.v, (short)0, cv[1][1], false, false);
        cg[0][0] = __builtin_amdgcn_wmma_f32_16x16x32_bf16(false, a0.v, false, bg0.v, (short)0, cg[0][0], false, false);
        cg[0][1] = __builtin_amdgcn_wmma_f32_16x16x32_bf16(false, a0.v, false, bg1.v, (short)0, cg[0][1], false, false);
        cg[1][0] = __builtin_amdgcn_wmma_f32_16x16x32_bf16(false, a1.v, false, bg0.v, (short)0, cg[1][0], false, false);
        cg[1][1] = __builtin_amdgcn_wmma_f32_16x16x32_bf16(false, a1.v, false, bg1.v, (short)0, cg[1][1], false, false);
    }

    // C/D layout: VGPR r -> row (tile_m + r + 8h), col (tile_n + l15).
    // V and G tiles share the layout, so the gate fuse is pure VALU.
    #pragma unroll
    for (int m = 0; m < 2; ++m)
        #pragma unroll
        for (int n = 0; n < 2; ++n)
            #pragma unroll
            for (int r = 0; r < 8; ++r) {
                float g   = cg[m][n][r];
                float val = cv[m][n][r] * (g / (1.0f + __expf(-g)));  // V * silu(G)
                int row   = mW + m * 16 + r + 8 * h;
                int col   = nW + n * 16 + l15;
                Gout[(size_t)row * DMODEL + col] = (bf16_t)val;
            }
}

// ---------------------------------------------------------------------------
// Kernel 3: proj = gated*Wo^T ; pre = proj + x ; out = LayerNorm(pre)*gamma+beta
// Block = 512 threads = 16 waves; block owns 32 rows x all 1024 cols so LN is
// fused. Each wave: 32x64 slab = 2 A-frags x 4 B-tiles -> 8 WMMAs / 12 loads.
// ---------------------------------------------------------------------------
__global__ __launch_bounds__(512) void out_ln_kernel(
    const bf16_t* __restrict__ Gb,    // [RTOTAL, 1024] gated bf16
    const bf16_t* __restrict__ Wo,    // [1024, 1024]  (W_O[d, v])
    const float*  __restrict__ X,     // [RTOTAL, 1024] fp32 residual
    const float*  __restrict__ gamma,
    const float*  __restrict__ beta,
    float* __restrict__ Out)          // [RTOTAL, 1024]
{
    __shared__ float redS[32];
    __shared__ float redQ[32];

    const int lane  = threadIdx.x & 31;
    const int wave  = threadIdx.x >> 5;   // 0..15
    const int h     = lane >> 4;
    const int l15   = lane & 15;
    const int mBase = blockIdx.x * 32;
    const int nW    = wave * 64;

    const bf16_t* arow0 = Gb + (size_t)(mBase + l15) * DMODEL;
    const bf16_t* arow1 = Gb + (size_t)(mBase + 16 + l15) * DMODEL;

    v8f zero = {0.f,0.f,0.f,0.f,0.f,0.f,0.f,0.f};
    v8f acc[2][4];
    #pragma unroll
    for (int m = 0; m < 2; ++m)
        #pragma unroll
        for (int t = 0; t < 4; ++t) acc[m][t] = zero;

    for (int k0 = 0; k0 < DMODEL; k0 += 32) {
        ABfrag a0, a1;
        a0.h[0] = *(const v8bf*)(arow0 + k0 + 16 * h);
        a0.h[1] = *(const v8bf*)(arow0 + k0 + 16 * h + 8);
        a1.h[0] = *(const v8bf*)(arow1 + k0 + 16 * h);
        a1.h[1] = *(const v8bf*)(arow1 + k0 + 16 * h + 8);
        #pragma unroll
        for (int t = 0; t < 4; ++t) {
            const bf16_t* wrow = Wo + (size_t)(nW + t * 16 + l15) * DMODEL;
            ABfrag b;
            b.h[0] = *(const v8bf*)(wrow + k0 + 16 * h);
            b.h[1] = *(const v8bf*)(wrow + k0 + 16 * h + 8);
            acc[0][t] = __builtin_amdgcn_wmma_f32_16x16x32_bf16(false, a0.v, false, b.v, (short)0, acc[0][t], false, false);
            acc[1][t] = __builtin_amdgcn_wmma_f32_16x16x32_bf16(false, a1.v, false, b.v, (short)0, acc[1][t], false, false);
        }
    }

    // Residual add (fp32 x)
    #pragma unroll
    for (int m = 0; m < 2; ++m)
        #pragma unroll
        for (int t = 0; t < 4; ++t) {
            int col = nW + t * 16 + l15;
            #pragma unroll
            for (int r = 0; r < 8; ++r) {
                int row = mBase + m * 16 + r + 8 * h;
                acc[m][t][r] += X[(size_t)row * DMODEL + col];
            }
        }

    if (threadIdx.x < 32) { redS[threadIdx.x] = 0.f; redQ[threadIdx.x] = 0.f; }
    __syncthreads();

    // Per-lane partials: each lane holds 4 cols of 16 rows (rows m*16 + r + 8h)
    #pragma unroll
    for (int m = 0; m < 2; ++m)
        #pragma unroll
        for (int r = 0; r < 8; ++r) {
            float s = 0.f, q = 0.f;
            #pragma unroll
            for (int t = 0; t < 4; ++t) { float v = acc[m][t][r]; s += v; q += v * v; }
            int rr = m * 16 + r + 8 * h;
            atomicAdd(&redS[rr], s);   // ds_add_f32
            atomicAdd(&redQ[rr], q);
        }
    __syncthreads();

    #pragma unroll
    for (int m = 0; m < 2; ++m)
        #pragma unroll
        for (int r = 0; r < 8; ++r) {
            int rr     = m * 16 + r + 8 * h;
            float mean = redS[rr] * (1.0f / 1024.0f);
            float var  = redQ[rr] * (1.0f / 1024.0f) - mean * mean;
            float rstd = rsqrtf(var + 1e-5f);
            int row    = mBase + rr;
            #pragma unroll
            for (int t = 0; t < 4; ++t) {
                int col = nW + t * 16 + l15;
                Out[(size_t)row * DMODEL + col] =
                    (acc[m][t][r] - mean) * rstd * gamma[col] + beta[col];
            }
        }
}

// ---------------------------------------------------------------------------
// Launch. Inputs (setup_inputs order): 0=x 1=W_Q 2=W_K 3=W_V 4=W_g 5=W_alpha
// 6=W_O 7=ln_gamma 8=ln_beta.  Q/K/alpha are dead code in the reference.
// Workspace layout (~70 MB): Xb | Gb | Wv_b | Wg_b | Wo_b (all bf16).
// ---------------------------------------------------------------------------
extern "C" void kernel_launch(void* const* d_in, const int* in_sizes, int n_in,
                              void* d_out, int out_size, void* d_ws, size_t ws_size,
                              hipStream_t stream) {
    (void)in_sizes; (void)n_in; (void)out_size; (void)ws_size;

    const float* x     = (const float*)d_in[0];
    const float* W_V   = (const float*)d_in[3];
    const float* W_g   = (const float*)d_in[4];
    const float* W_O   = (const float*)d_in[6];
    const float* gamma = (const float*)d_in[7];
    const float* beta  = (const float*)d_in[8];
    float* out = (float*)d_out;

    char* ws = (char*)d_ws;
    bf16_t* Xb  = (bf16_t*)ws;  ws += (size_t)RTOTAL * DMODEL * 2;
    bf16_t* Gb  = (bf16_t*)ws;  ws += (size_t)RTOTAL * DMODEL * 2;
    bf16_t* Wvb = (bf16_t*)ws;  ws += (size_t)DMODEL * DMODEL * 2;
    bf16_t* Wgb = (bf16_t*)ws;  ws += (size_t)DMODEL * DMODEL * 2;
    bf16_t* Wob = (bf16_t*)ws;

    cvt_f32_bf16<<<4096, 256, 0, stream>>>(x,   Xb,  RTOTAL * DMODEL);
    cvt_f32_bf16<<<1024, 256, 0, stream>>>(W_V, Wvb, DMODEL * DMODEL);
    cvt_f32_bf16<<<1024, 256, 0, stream>>>(W_g, Wgb, DMODEL * DMODEL);
    cvt_f32_bf16<<<1024, 256, 0, stream>>>(W_O, Wob, DMODEL * DMODEL);

    // grid: x = N tiles (1024/64), y = M tiles (16384/128)
    proj_gate_kernel<<<dim3(16, 128), 256, 0, stream>>>(Xb, Wvb, Wgb, Gb);
    // one block per 32 rows, full width for fused LayerNorm
    out_ln_kernel<<<RTOTAL / 32, 512, 0, stream>>>(Gb, Wob, x, gamma, beta, out);
}
